// MoELayer_14998025797648
// MI455X (gfx1250) — matrix operand order repaired and checked by reading, not compiled, not run on hardware
//
#include <hip/hip_runtime.h>
#include <hip/hip_bf16.h>

typedef __bf16 bf16_t;
typedef __attribute__((ext_vector_type(16))) __bf16 v16bf;
typedef __attribute__((ext_vector_type(8)))  __bf16 v8bf;
typedef __attribute__((ext_vector_type(2)))  __bf16 v2bf;
typedef __attribute__((ext_vector_type(8)))  float  v8f;

#define T_TOK 8192
#define DIM   1024
#define HID   2816
#define NEXP  8
#define KACT  2
#define MROWS 64
#define MT64  (T_TOK / MROWS)    // 128
#define LDA   (DIM + 8)          // padded LDS row stride, rows stay 16B aligned

#define WMMA(a, b, c) \
  (c) = __builtin_amdgcn_wmma_f32_16x16x32_bf16(false, (a), false, (b), (short)0, (c), false, false)

// ---------------------------------------------------------------- init ------
__global__ void init_kernel(int* __restrict__ tcnt, int* __restrict__ counts,
                            float* __restrict__ sumP) {
  int i = blockIdx.x * blockDim.x + threadIdx.x;
  if (i < T_TOK) tcnt[i] = 0;
  if (i < NEXP) { counts[i] = 0; sumP[i] = 0.f; }
}

// -------------------------------------------------------------- router ------
__global__ void router_kernel(const float* __restrict__ x, const float* __restrict__ Wr,
                              int* __restrict__ counts, float* __restrict__ sumP,
                              int* __restrict__ lists, float* __restrict__ wlist) {
  __shared__ float sP[NEXP];
  int tid = threadIdx.x;
  if (tid < NEXP) sP[tid] = 0.f;
  __syncthreads();

  int wave = tid >> 5, lane = tid & 31;
  int t = blockIdx.x * 8 + wave;

  float acc[NEXP];
#pragma unroll
  for (int e = 0; e < NEXP; ++e) acc[e] = 0.f;
  const float* xr = x + (size_t)t * DIM;
  for (int d = lane; d < DIM; d += 32) {
    float xv = xr[d];
#pragma unroll
    for (int e = 0; e < NEXP; ++e) acc[e] += xv * Wr[d * NEXP + e];
  }
#pragma unroll
  for (int e = 0; e < NEXP; ++e) {
#pragma unroll
    for (int off = 16; off > 0; off >>= 1) acc[e] += __shfl_xor(acc[e], off, 32);
  }
  float mx = acc[0];
#pragma unroll
  for (int e = 1; e < NEXP; ++e) mx = fmaxf(mx, acc[e]);
  float p[NEXP], s = 0.f;
#pragma unroll
  for (int e = 0; e < NEXP; ++e) { p[e] = __expf(acc[e] - mx); s += p[e]; }
  float inv = __builtin_amdgcn_rcpf(s);
#pragma unroll
  for (int e = 0; e < NEXP; ++e) p[e] *= inv;
  int i0 = 0;
#pragma unroll
  for (int e = 1; e < NEXP; ++e) if (p[e] > p[i0]) i0 = e;
  int i1 = (i0 == 0) ? 1 : 0;
#pragma unroll
  for (int e = 0; e < NEXP; ++e) if (e != i0 && p[e] > p[i1]) i1 = e;
  float wsum = p[i0] + p[i1];
  float w0 = p[i0] / wsum, w1 = p[i1] / wsum;

  if (lane == 0) {
#pragma unroll
    for (int e = 0; e < NEXP; ++e) atomicAdd(&sP[e], p[e]);
    int pos0 = atomicAdd(&counts[i0], 1);
    lists[i0 * T_TOK + pos0] = t;  wlist[i0 * T_TOK + pos0] = w0;
    int pos1 = atomicAdd(&counts[i1], 1);
    lists[i1 * T_TOK + pos1] = t;  wlist[i1 * T_TOK + pos1] = w1;
  }
  __syncthreads();
  if (tid < NEXP) atomicAdd(&sumP[tid], sP[tid]);
}

// ------------------------------------------------- scan + compact -----------
__global__ void scan_kernel(const int* __restrict__ counts, int* __restrict__ offsets) {
  if (threadIdx.x == 0 && blockIdx.x == 0) {
    int s = 0;
#pragma unroll
    for (int e = 0; e < NEXP; ++e) { offsets[e] = s; s += counts[e]; }
    offsets[NEXP] = s;
  }
}

__global__ void compact_kernel(const int* __restrict__ counts, const int* __restrict__ offsets,
                               const int* __restrict__ lists, const float* __restrict__ wlist,
                               float* __restrict__ rowW, int* __restrict__ tokSlot,
                               int* __restrict__ tcnt) {
  int e = blockIdx.y;
  int i = blockIdx.x * blockDim.x + threadIdx.x;
  if (i < counts[e]) {
    int s = offsets[e] + i;
    int t = lists[e * T_TOK + i];
    rowW[s] = wlist[e * T_TOK + i];
    int k = atomicAdd(&tcnt[t], 1);          // slot order irrelevant (commutative sum)
    tokSlot[2 * t + k] = s;
  }
}

// ------------------------------------- weight fp32 -> swizzled bf16 ---------
// 32x16 (KxN) tile -> 32 lanes x 16 contiguous bf16; tile order K-inner.
__global__ void swz_kernel(const float* __restrict__ W, bf16_t* __restrict__ out,
                           int Kt, int Nt) {
  size_t total = (size_t)NEXP * Kt * Nt / 8;
  size_t gid = (size_t)blockIdx.x * blockDim.x + threadIdx.x;
  size_t gstride = (size_t)gridDim.x * blockDim.x;
  int numKT = Kt / 32;
  int tilesPer = numKT * (Nt / 16);
  for (; gid < total; gid += gstride) {
    int half = (int)(gid & 1);
    int lane = (int)((gid >> 1) & 31);
    size_t tileG = gid >> 6;
    int ex   = (int)(tileG / tilesPer);
    int tile = (int)(tileG % tilesPer);
    int nt = tile / numKT, kt = tile % numKT;
    int n  = nt * 16 + (lane & 15);
    int k0 = kt * 32 + 8 * (lane >> 4) + 16 * half;
    const float* Wp = W + (size_t)ex * Kt * Nt;
    bf16_t* op = out + (size_t)ex * Kt * Nt + (size_t)tile * 512 + lane * 16 + half * 8;
    v8bf v;
#pragma unroll
    for (int i = 0; i < 8; ++i) v[i] = (bf16_t)Wp[(size_t)(k0 + i) * Nt + n];
    *reinterpret_cast<v8bf*>(op) = v;
  }
}

// ---------------------------------------------------- fragment loaders ------
__device__ __forceinline__ v16bf load_a_lds(const bf16_t* base, int ld, int kb, int lane) {
  int m = lane & 15, hi = lane >> 4;
  const bf16_t* p = base + m * ld + kb + 8 * hi;
  v8bf lo = *reinterpret_cast<const v8bf*>(p);
  v8bf hh = *reinterpret_cast<const v8bf*>(p + 16);
  v16bf a;
#pragma unroll
  for (int i = 0; i < 8; ++i) { a[i] = lo[i]; a[i + 8] = hh[i]; }
  return a;
}

__device__ __forceinline__ v16bf load_a_glob(const bf16_t* __restrict__ rowbase, int ld,
                                             int kb, int lane) {
  int m = lane & 15, hi = lane >> 4;
  const bf16_t* p = rowbase + (size_t)m * ld + kb + 8 * hi;
  v8bf lo = *reinterpret_cast<const v8bf*>(p);
  v8bf hh = *reinterpret_cast<const v8bf*>(p + 16);
  v16bf a;
#pragma unroll
  for (int i = 0; i < 8; ++i) { a[i] = lo[i]; a[i + 8] = hh[i]; }
  return a;
}

__device__ __forceinline__ v16bf load_b_swz(const bf16_t* __restrict__ base, int numKT,
                                            int kt, int nt, int lane) {
  const bf16_t* p = base + ((size_t)(nt * numKT + kt)) * 512 + lane * 16;
  v8bf lo = *reinterpret_cast<const v8bf*>(p);
  v8bf hh = *reinterpret_cast<const v8bf*>(p + 8);
  v16bf b;
#pragma unroll
  for (int i = 0; i < 8; ++i) { b[i] = lo[i]; b[i + 8] = hh[i]; }
  return b;
}

// silu(g1)*g3 epilogue for one 16x16 C tile; FULL=true -> no row guards
template <bool FULL>
__device__ __forceinline__ void h_store(v8f g1, v8f g3, int mbase, int rv,
                                        bf16_t* __restrict__ hb, size_t rowBase,
                                        int nb, int lane) {
  int hi = lane >> 4, n = lane & 15;
#pragma unroll
  for (int j = 0; j < 8; ++j) {
    int m = mbase + j + 8 * hi;
    float g = g1[j];
    float hv = g * __builtin_amdgcn_rcpf(1.f + __expf(-g)) * g3[j];
    if (FULL || m < rv) hb[(rowBase + m) * HID + nb + n] = (bf16_t)hv;
  }
}

template <bool FULL>
__device__ __forceinline__ void y_store(v8f c, int mbase, int rv,
                                        float* __restrict__ yb, size_t rowBase,
                                        int nb, int lane) {
  int hi = lane >> 4, n = lane & 15;
#pragma unroll
  for (int j = 0; j < 8; ++j) {
    int m = mbase + j + 8 * hi;
    if (FULL || m < rv) yb[(rowBase + m) * DIM + nb + n] = c[j];
  }
}

// ------------------------------------ phase 1: h = silu(x@w1)*(x@w3) --------
__global__ void h_kernel(const float* __restrict__ x,
                         const bf16_t* __restrict__ w1s, const bf16_t* __restrict__ w3s,
                         const int* __restrict__ counts, const int* __restrict__ offsets,
                         const int* __restrict__ lists, bf16_t* __restrict__ hbuf) {
  extern __shared__ char smem[];
  bf16_t* xA  = (bf16_t*)smem;              // 64 x LDA bf16 (132 KB)
  int*    tokS = (int*)(xA + MROWS * LDA);

  int e  = blockIdx.x >> 7;
  int mt = blockIdx.x & (MT64 - 1);
  int ne = counts[e];
  int row0 = mt * MROWS;
  if (row0 >= ne) return;
  int rv = ne - row0; if (rv > MROWS) rv = MROWS;

  int tid = threadIdx.x, wave = tid >> 5, lane = tid & 31;
  if (tid < MROWS) tokS[tid] = (tid < rv) ? lists[e * T_TOK + row0 + tid] : 0;
  __syncthreads();

  // gather 64 token rows, fp32 -> packed bf16 pairs (ds_store_b32)
  for (int i = tid; i < MROWS * (DIM / 2); i += 256) {
    int r = i >> 9, c2 = (i & 511) * 2;
    float v0 = 0.f, v1 = 0.f;
    if (r < rv) {
      const float* xp = x + (size_t)tokS[r] * DIM + c2;
      v0 = xp[0]; v1 = xp[1];
    }
    v2bf pv; pv[0] = (bf16_t)v0; pv[1] = (bf16_t)v1;
    *reinterpret_cast<v2bf*>(&xA[r * LDA + c2]) = pv;
  }
  __syncthreads();

  const bf16_t* w1e = w1s + (size_t)e * DIM * HID;
  const bf16_t* w3e = w3s + (size_t)e * DIM * HID;
  size_t rowBase = (size_t)offsets[e] + row0;
  const int numKT = DIM / 32;   // 32

  for (int nt = wave; nt < HID / 16; nt += 8) {
    v8f c10 = {}, c11 = {}, c12 = {}, c13 = {};
    v8f c30 = {}, c31 = {}, c32 = {}, c33 = {};
#pragma unroll 1
    for (int kt = 0; kt < numKT; ++kt) {
      v16bf b1 = load_b_swz(w1e, numKT, kt, nt, lane);
      v16bf b3 = load_b_swz(w3e, numKT, kt, nt, lane);
      __builtin_prefetch(w1e + ((size_t)(nt * numKT + kt + 1)) * 512 + lane * 16, 0, 3);
      __builtin_prefetch(w3e + ((size_t)(nt * numKT + kt + 1)) * 512 + lane * 16, 0, 3);
      int kb = kt * 32;
      v16bf a0 = load_a_lds(xA,             LDA, kb, lane);
      WMMA(a0, b1, c10); WMMA(a0, b3, c30);
      v16bf a1 = load_a_lds(xA + 16 * LDA,  LDA, kb, lane);
      WMMA(a1, b1, c11); WMMA(a1, b3, c31);
      v16bf a2 = load_a_lds(xA + 32 * LDA,  LDA, kb, lane);
      WMMA(a2, b1, c12); WMMA(a2, b3, c32);
      v16bf a3 = load_a_lds(xA + 48 * LDA,  LDA, kb, lane);
      WMMA(a3, b1, c13); WMMA(a3, b3, c33);
    }
    int nb = nt * 16;
    if (rv == MROWS) {          // hot path: no row guards, no EXEC churn
      h_store<true>(c10, c30,  0, rv, hbuf, rowBase, nb, lane);
      h_store<true>(c11, c31, 16, rv, hbuf, rowBase, nb, lane);
      h_store<true>(c12, c32, 32, rv, hbuf, rowBase, nb, lane);
      h_store<true>(c13, c33, 48, rv, hbuf, rowBase, nb, lane);
    } else {                    // ragged last tile of the expert segment
      h_store<false>(c10, c30,  0, rv, hbuf, rowBase, nb, lane);
      h_store<false>(c11, c31, 16, rv, hbuf, rowBase, nb, lane);
      h_store<false>(c12, c32, 32, rv, hbuf, rowBase, nb, lane);
      h_store<false>(c13, c33, 48, rv, hbuf, rowBase, nb, lane);
    }
  }
}

// ------------------------------------ phase 2: ybuf[slot] = h @ w2 ----------
__global__ void y_kernel(const bf16_t* __restrict__ hbuf, const bf16_t* __restrict__ w2s,
                         const int* __restrict__ counts, const int* __restrict__ offsets,
                         float* __restrict__ ybuf) {
  int e  = blockIdx.x >> 7;
  int mt = blockIdx.x & (MT64 - 1);
  int ne = counts[e];
  int row0 = mt * MROWS;
  if (row0 >= ne) return;
  int rv = ne - row0; if (rv > MROWS) rv = MROWS;
  size_t rowBase = (size_t)offsets[e] + row0;

  int tid = threadIdx.x, wave = tid >> 5, lane = tid & 31;
  const bf16_t* w2e  = w2s + (size_t)e * HID * DIM;
  const bf16_t* hrow = hbuf + rowBase * HID;
  const int numKT = HID / 32;   // 88

  for (int nt = wave; nt < DIM / 16; nt += 8) {
    v8f c0 = {}, c1 = {}, c2 = {}, c3 = {};
#pragma unroll 1
    for (int kt = 0; kt < numKT; ++kt) {
      v16bf b = load_b_swz(w2e, numKT, kt, nt, lane);
      __builtin_prefetch(w2e + ((size_t)(nt * numKT + kt + 1)) * 512 + lane * 16, 0, 3);
      int kb = kt * 32;
      v16bf a0 = load_a_glob(hrow,                      HID, kb, lane);
      WMMA(a0, b, c0);
      v16bf a1 = load_a_glob(hrow + (size_t)16 * HID,   HID, kb, lane);
      WMMA(a1, b, c1);
      v16bf a2 = load_a_glob(hrow + (size_t)32 * HID,   HID, kb, lane);
      WMMA(a2, b, c2);
      v16bf a3 = load_a_glob(hrow + (size_t)48 * HID,   HID, kb, lane);
      WMMA(a3, b, c3);
    }
    int nb = nt * 16;
    if (rv == MROWS) {
      y_store<true>(c0,  0, rv, ybuf, rowBase, nb, lane);
      y_store<true>(c1, 16, rv, ybuf, rowBase, nb, lane);
      y_store<true>(c2, 32, rv, ybuf, rowBase, nb, lane);
      y_store<true>(c3, 48, rv, ybuf, rowBase, nb, lane);
    } else {
      y_store<false>(c0,  0, rv, ybuf, rowBase, nb, lane);
      y_store<false>(c1, 16, rv, ybuf, rowBase, nb, lane);
      y_store<false>(c2, 32, rv, ybuf, rowBase, nb, lane);
      y_store<false>(c3, 48, rv, ybuf, rowBase, nb, lane);
    }
  }
}

// ------------------------ combine: out[t] = w0*y[s0] + w1*y[s1] -------------
__global__ void combine_kernel(const float* __restrict__ ybuf, const int* __restrict__ tokSlot,
                               const float* __restrict__ rowW, float* __restrict__ out) {
  int t = blockIdx.x;
  int s0 = tokSlot[2 * t], s1 = tokSlot[2 * t + 1];
  float w0 = rowW[s0], w1 = rowW[s1];
  const float4* y0 = (const float4*)(ybuf + (size_t)s0 * DIM);
  const float4* y1 = (const float4*)(ybuf + (size_t)s1 * DIM);
  float4* op = (float4*)(out + (size_t)t * DIM);
  int i = threadIdx.x;                 // 256 threads == DIM/4
  float4 a = y0[i], b = y1[i];
  float4 r;
  r.x = w0 * a.x + w1 * b.x;  r.y = w0 * a.y + w1 * b.y;
  r.z = w0 * a.z + w1 * b.z;  r.w = w0 * a.w + w1 * b.w;
  op[i] = r;
}

// ----------------------------------------------------------------- aux ------
__global__ void aux_kernel(const int* __restrict__ counts, const float* __restrict__ sumP,
                           float* __restrict__ out_aux) {
  if (threadIdx.x == 0 && blockIdx.x == 0) {
    float a = 0.f;
#pragma unroll
    for (int e = 0; e < NEXP; ++e) {
      float f = (float)counts[e] / (float)(T_TOK * KACT);
      float P = sumP[e] / (float)T_TOK;
      a += f * P;
    }
    out_aux[0] = (float)NEXP * a;
  }
}

// -------------------------------------------------------------- launch ------
extern "C" void kernel_launch(void* const* d_in, const int* in_sizes, int n_in,
                              void* d_out, int out_size, void* d_ws, size_t ws_size,
                              hipStream_t stream) {
  (void)in_sizes; (void)n_in; (void)out_size; (void)ws_size;
  const float* x  = (const float*)d_in[0];
  const float* Wr = (const float*)d_in[1];
  const float* w1 = (const float*)d_in[2];
  const float* w3 = (const float*)d_in[3];
  const float* w2 = (const float*)d_in[4];
  float* out = (float*)d_out;

  char* wp = (char*)d_ws;
  int*    counts  = (int*)(wp + 0);
  int*    offsets = (int*)(wp + 64);
  float*  sumP    = (float*)(wp + 128);
  wp += 256;
  int*    lists   = (int*)wp;   wp += sizeof(int)   * NEXP * T_TOK;   // 256 KB
  float*  wlist   = (float*)wp; wp += sizeof(float) * NEXP * T_TOK;   // 256 KB
  float*  rowW    = (float*)wp; wp += sizeof(float) * KACT * T_TOK;   // 64 KB
  int*    tokSlot = (int*)wp;   wp += sizeof(int)   * KACT * T_TOK;   // 64 KB
  int*    tcnt    = (int*)wp;   wp += sizeof(int)   * T_TOK;          // 32 KB
  wp = (char*)(((size_t)wp + 255) & ~(size_t)255);
  bf16_t* w1s = (bf16_t*)wp;    wp += sizeof(bf16_t) * (size_t)NEXP * DIM * HID;
  bf16_t* w3s = (bf16_t*)wp;    wp += sizeof(bf16_t) * (size_t)NEXP * DIM * HID;
  bf16_t* w2s = (bf16_t*)wp;    wp += sizeof(bf16_t) * (size_t)NEXP * HID * DIM;
  bf16_t* hbuf = (bf16_t*)wp;   wp += sizeof(bf16_t) * ((size_t)KACT * T_TOK + MROWS) * HID;
  float*  ybuf = (float*)wp;    // KACT*T_TOK x DIM f32 (67 MB)

  init_kernel<<<(T_TOK + 255) / 256, 256, 0, stream>>>(tcnt, counts, sumP);
  router_kernel<<<T_TOK / 8, 256, 0, stream>>>(x, Wr, counts, sumP, lists, wlist);
  scan_kernel<<<1, 32, 0, stream>>>(counts, offsets);
  compact_kernel<<<dim3(T_TOK / 256, NEXP), 256, 0, stream>>>(counts, offsets, lists, wlist,
                                                              rowW, tokSlot, tcnt);
  swz_kernel<<<8192, 256, 0, stream>>>(w1, w1s, DIM, HID);
  swz_kernel<<<8192, 256, 0, stream>>>(w3, w3s, DIM, HID);
  swz_kernel<<<8192, 256, 0, stream>>>(w2, w2s, HID, DIM);

  size_t shmem_h = sizeof(bf16_t) * MROWS * LDA + sizeof(int) * MROWS;
  h_kernel<<<NEXP * MT64, 256, shmem_h, stream>>>(x, w1s, w3s, counts, offsets, lists, hbuf);
  y_kernel<<<NEXP * MT64, 256, 0, stream>>>(hbuf, w2s, counts, offsets, ybuf);
  combine_kernel<<<T_TOK, 256, 0, stream>>>(ybuf, tokSlot, rowW, out);
  aux_kernel<<<1, 32, 0, stream>>>(counts, sumP, out + (size_t)T_TOK * DIM);
}
